// CodebookDecoder_87428354277942
// MI455X (gfx1250) — matrix operand order repaired and checked by reading, not compiled, not run on hardware
//
#include <hip/hip_runtime.h>
#include <hip/hip_bf16.h>
#include <math.h>
#include <cstddef>
#include <cstdint>

// ---------------------------------------------------------------------------
// Types / helpers
// ---------------------------------------------------------------------------
typedef __bf16 bf16_t;
typedef __bf16 v16bf  __attribute__((ext_vector_type(16)));
typedef __bf16 bf16x8 __attribute__((ext_vector_type(8)));
typedef float  v8f    __attribute__((ext_vector_type(8)));
typedef float  f32x4  __attribute__((ext_vector_type(4)));
typedef int    i32x4  __attribute__((ext_vector_type(4)));

#define DEV static __device__ __forceinline__

// CDNA5 async global->LDS copy path (gated: probe via __has_builtin)
#if defined(__has_builtin)
#if __has_builtin(__builtin_amdgcn_global_load_async_to_lds_b128) && \
    __has_builtin(__builtin_amdgcn_s_wait_asynccnt)
#define USE_ASYNC_LDS 1
#endif
#endif
#ifndef USE_ASYNC_LDS
#define USE_ASYNC_LDS 0
#endif

#if USE_ASYNC_LDS
DEV void async_cp16(const void* g, void* l) {
  // 16 bytes per lane, ASYNCcnt-tracked, writes LDS directly.
  // Builtin prototype: (int4 AS1*, int4 AS3*, imm offset, imm cpol)
  auto gp = (__attribute__((address_space(1))) i32x4*)(uintptr_t)g;
  auto lp = (__attribute__((address_space(3))) i32x4*)(uint32_t)(uintptr_t)l;
  __builtin_amdgcn_global_load_async_to_lds_b128(gp, lp, 0, 0);
}
DEV void wait_async0() { __builtin_amdgcn_s_wait_asynccnt(0); }
#endif

DEV bf16_t f2bf(float x) {
  unsigned u = __builtin_bit_cast(unsigned, x);
  unsigned r = (u + 0x7FFFu + ((u >> 16) & 1u)) >> 16;   // round-nearest-even
  unsigned short s = (unsigned short)r;
  return __builtin_bit_cast(bf16_t, s);
}
DEV float gelu_exact(float x) {
  return 0.5f * x * (1.0f + erff(x * 0.70710678118654752f));
}

// Load one 32x16 B fragment from an LDS row-major tile (row stride KP bf16)
DEV v16bf load_bfrag(const bf16_t* base, int ks, int hf) {
  const bf16_t* bp = base + ks * 32 + hf * 8;
  bf16x8 b0 = *(const bf16x8*)bp;
  bf16x8 b1 = *(const bf16x8*)(bp + 16);
  v16bf b;
#pragma unroll
  for (int i = 0; i < 8; ++i) { b[i] = b0[i]; b[8 + i] = b1[i]; }
  return b;
}

// Problem constants
constexpr int TOKN = 34, TD = 512, KCB = 2048, JJ = 17, LL = 4;
constexpr int HH = 256, HID = 512, TI = 64, BS = 2048;
constexpr int NE    = BS * TOKN;   // 69632 encode rows
constexpr int MROWS = BS * JJ;     // 34816 decoder rows

// ---------------------------------------------------------------------------
// Utility kernels
// ---------------------------------------------------------------------------
__global__ void zero4_kernel(float* __restrict__ p, long long n4) {
  long long i = (long long)blockIdx.x * blockDim.x + threadIdx.x;
  long long stride = (long long)gridDim.x * blockDim.x;
  f32x4 z = {0.f, 0.f, 0.f, 0.f};
  for (; i < n4; i += stride) ((f32x4*)p)[i] = z;
}

__global__ void cvt_bf16_kernel(const float* __restrict__ in, bf16_t* __restrict__ out, int n) {
  int i = blockIdx.x * blockDim.x + threadIdx.x;
  int stride = gridDim.x * blockDim.x;
  for (; i < n; i += stride) out[i] = f2bf(in[i]);
}

__global__ __launch_bounds__(256) void cnorm_kernel(const float* __restrict__ C,
                                                    float* __restrict__ out) {
  const int w = threadIdx.x >> 5, lane = threadIdx.x & 31;
  const int row = blockIdx.x * 8 + w;
  const float* cr = C + (size_t)row * TD;
  float s = 0.f;
#pragma unroll
  for (int i = 0; i < 16; ++i) { float x = cr[lane + i * 32]; s += x * x; }
#pragma unroll
  for (int off = 1; off < 32; off <<= 1) s += __shfl_xor(s, off, 32);
  if (lane == 0) out[row] = s;
}

// ---------------------------------------------------------------------------
// Quantization: WMMA bf16 E·C^T, fused argmin(||c||^2 - 2 e·c), gather + onehot
// Double-buffered async LDS staging; dual accumulator chains.
// ---------------------------------------------------------------------------
__global__ __launch_bounds__(256) void quant_kernel(
    const float* __restrict__ E, const bf16_t* __restrict__ CBbf,
    const float* __restrict__ cnorm, const float* __restrict__ CBf32,
    float* __restrict__ orig, float* __restrict__ enc, float* __restrict__ idxout) {
#if USE_ASYNC_LDS
  __shared__ bf16_t cbt[2][16 * 520];
#else
  __shared__ bf16_t cbt1[16 * 520];
#endif
  __shared__ int bidx_s[128];

  const int t = threadIdx.x;
  const int w = t >> 5, lane = t & 31, hf = lane >> 4, lsub = lane & 15;
  const int mblk = blockIdx.x * 128;
  const int m0 = mblk + w * 16;

  // A fragments: this lane's E row, all 16 k-steps (K=512), f32->bf16 inline.
  v16bf afr[16];
  const float* arow = E + (size_t)(m0 + lsub) * TD;
#pragma unroll
  for (int ks = 0; ks < 16; ++ks) {
    const int kb = ks * 32 + hf * 8;
    f32x4 p0 = *(const f32x4*)(arow + kb);
    f32x4 p1 = *(const f32x4*)(arow + kb + 4);
    f32x4 p2 = *(const f32x4*)(arow + kb + 16);
    f32x4 p3 = *(const f32x4*)(arow + kb + 20);
    v16bf a;
#pragma unroll
    for (int i = 0; i < 4; ++i) {
      a[i] = f2bf(p0[i]); a[4 + i] = f2bf(p1[i]);
      a[8 + i] = f2bf(p2[i]); a[12 + i] = f2bf(p3[i]);
    }
    afr[ks] = a;
  }

  float best[8]; int bi[8];
#pragma unroll
  for (int r = 0; r < 8; ++r) { best[r] = 3.0e38f; bi[r] = 0; }

#if USE_ASYNC_LDS
  auto issue_tile = [&](int n0, int p) {
#pragma unroll
    for (int i = 0; i < 4; ++i) {
      int u8 = t + i * 256;            // unit of 8 bf16 (16B)
      int row = u8 >> 6;
      int col = (u8 & 63) << 3;
      async_cp16(CBbf + (size_t)(n0 + row) * TD + col, &cbt[p][row * 520 + col]);
    }
  };
  issue_tile(0, 0);
  int p = 0;
#endif

  for (int n0 = 0; n0 < KCB; n0 += 16) {
#if USE_ASYNC_LDS
    wait_async0();                      // tile for this iteration has landed
    __syncthreads();                    // visible to all waves; prior reads done
    if (n0 + 16 < KCB) issue_tile(n0 + 16, p ^ 1);
    const bf16_t* base = &cbt[p][lsub * 520];
#else
    __syncthreads();
#pragma unroll
    for (int i = 0; i < 4; ++i) {
      int u8 = t + i * 256;
      int row = u8 >> 6;
      int col = (u8 & 63) << 3;
      *(bf16x8*)&cbt1[row * 520 + col] =
          *(const bf16x8*)(CBbf + (size_t)(n0 + row) * TD + col);
    }
    __syncthreads();
    const bf16_t* base = &cbt1[lsub * 520];
#endif

    // two independent accumulator chains (even/odd k-steps)
    v8f acc0 = {0.f, 0.f, 0.f, 0.f, 0.f, 0.f, 0.f, 0.f};
    v8f acc1 = {0.f, 0.f, 0.f, 0.f, 0.f, 0.f, 0.f, 0.f};
#pragma unroll
    for (int ks = 0; ks < 16; ks += 2) {
      v16bf b0 = load_bfrag(base, ks, hf);
      v16bf b1 = load_bfrag(base, ks + 1, hf);
      acc0 = __builtin_amdgcn_wmma_f32_16x16x32_bf16(false, afr[ks], false, b0,
                                                     (short)0, acc0, false, false);
      acc1 = __builtin_amdgcn_wmma_f32_16x16x32_bf16(false, afr[ks + 1], false, b1,
                                                     (short)0, acc1, false, false);
    }
    v8f acc = acc0 + acc1;

    const float cv = cnorm[n0 + lsub];
    const int n = n0 + lsub;
#pragma unroll
    for (int r = 0; r < 8; ++r) {
      float d = cv - 2.0f * acc[r];     // ||e||^2 constant: irrelevant to argmin
      if (d < best[r]) { best[r] = d; bi[r] = n; }
    }
#if USE_ASYNC_LDS
    p ^= 1;
#endif
  }

  // reduce across the 16 lanes of each half (xor masks stay within half)
#pragma unroll
  for (int off = 1; off < 16; off <<= 1) {
#pragma unroll
    for (int r = 0; r < 8; ++r) {
      float od = __shfl_xor(best[r], off, 32);
      int   oi = __shfl_xor(bi[r],   off, 32);
      if (od < best[r] || (od == best[r] && oi < bi[r])) { best[r] = od; bi[r] = oi; }
    }
  }
  if (lsub == 0) {
#pragma unroll
    for (int r = 0; r < 8; ++r) bidx_s[w * 16 + hf * 8 + r] = bi[r];
  }
  __syncthreads();

  if (t < 128) {
    int id = bidx_s[t];
    idxout[mblk + t] = (float)id;
    enc[(size_t)(mblk + t) * KCB + id] = 1.0f;    // one-hot scatter (pre-zeroed)
  }
  // orig = e + (codebook[idx] - e)  (straight-through, matches fp32 rounding)
  for (int r = 0; r < 16; ++r) {
    const int row = m0 + r;
    const int id = bidx_s[w * 16 + r];
    const f32x4* src  = (const f32x4*)(CBf32 + (size_t)id * TD);
    const f32x4* esrc = (const f32x4*)(E + (size_t)row * TD);
    f32x4* dst = (f32x4*)(orig + (size_t)row * TD);
    for (int c = lane; c < 128; c += 32) {
      f32x4 ev = esrc[c];
      dst[c] = ev + (src[c] - ev);
    }
  }
}

// ---------------------------------------------------------------------------
// Generic bf16 WMMA GEMM: out[M,N] = act(A[M,KD] * B[N,KD]^T + bias) (+res)
// block = 256 thr / 8 waves covering 128 rows; async double-buffered B tiles.
// ---------------------------------------------------------------------------
template <int KD, bool GELU, bool RES, bool OBF>
__global__ __launch_bounds__(256) void gemm_wmma_kernel(
    const bf16_t* __restrict__ A, const bf16_t* __restrict__ B,
    const float* __restrict__ bias, const float* __restrict__ res,
    float* __restrict__ outF, bf16_t* __restrict__ outB, int N) {
  constexpr int KP = KD + 8;
  constexpr int KS = KD / 32;          // wmma k-steps (even)
  constexpr int CH = KD / 128;         // 16B copy chunks per thread per tile
#if USE_ASYNC_LDS
  __shared__ bf16_t bt[2][16 * KP];
#else
  __shared__ bf16_t bt1[16 * KP];
#endif

  const int t = threadIdx.x, w = t >> 5, lane = t & 31, hf = lane >> 4, lsub = lane & 15;
  const int m0 = blockIdx.x * 128 + w * 16;

  v16bf afr[KS];
  const bf16_t* arow = A + (size_t)(m0 + lsub) * KD;
#pragma unroll
  for (int ks = 0; ks < KS; ++ks) {
    const int kb = ks * 32 + hf * 8;
    bf16x8 p0 = *(const bf16x8*)(arow + kb);
    bf16x8 p1 = *(const bf16x8*)(arow + kb + 16);
    v16bf a;
#pragma unroll
    for (int i = 0; i < 8; ++i) { a[i] = p0[i]; a[8 + i] = p1[i]; }
    afr[ks] = a;
  }

#if USE_ASYNC_LDS
  auto issue_tile = [&](int n0, int p) {
#pragma unroll
    for (int i = 0; i < CH; ++i) {
      int u8 = t + i * 256;
      int row = u8 / (KD / 8);
      int col = (u8 % (KD / 8)) * 8;
      async_cp16(B + (size_t)(n0 + row) * KD + col, &bt[p][row * KP + col]);
    }
  };
  issue_tile(0, 0);
  int p = 0;
#endif

  for (int n0 = 0; n0 < N; n0 += 16) {
#if USE_ASYNC_LDS
    wait_async0();
    __syncthreads();
    if (n0 + 16 < N) issue_tile(n0 + 16, p ^ 1);
    const bf16_t* base = &bt[p][lsub * KP];
#else
    __syncthreads();
#pragma unroll
    for (int i = 0; i < CH; ++i) {
      int u8 = t + i * 256;
      int row = u8 / (KD / 8);
      int col = (u8 % (KD / 8)) * 8;
      *(bf16x8*)&bt1[row * KP + col] =
          *(const bf16x8*)(B + (size_t)(n0 + row) * KD + col);
    }
    __syncthreads();
    const bf16_t* base = &bt1[lsub * KP];
#endif

    // two independent accumulator chains (even/odd k-steps)
    v8f acc0 = {0.f, 0.f, 0.f, 0.f, 0.f, 0.f, 0.f, 0.f};
    v8f acc1 = {0.f, 0.f, 0.f, 0.f, 0.f, 0.f, 0.f, 0.f};
#pragma unroll
    for (int ks = 0; ks < KS; ks += 2) {
      v16bf b0 = load_bfrag(base, ks, hf);
      v16bf b1 = load_bfrag(base, ks + 1, hf);
      acc0 = __builtin_amdgcn_wmma_f32_16x16x32_bf16(false, afr[ks], false, b0,
                                                     (short)0, acc0, false, false);
      acc1 = __builtin_amdgcn_wmma_f32_16x16x32_bf16(false, afr[ks + 1], false, b1,
                                                     (short)0, acc1, false, false);
    }
    v8f acc = acc0 + acc1;

    const int col = n0 + lsub;
    const float bv = bias[col];
#pragma unroll
    for (int r = 0; r < 8; ++r) {
      const int row = m0 + hf * 8 + r;
      float v = acc[r] + bv;
      if (GELU) v = gelu_exact(v);
      if (RES)  v += res[(size_t)row * N + col];
      if (OBF)  outB[(size_t)row * N + col] = f2bf(v);
      else      outF[(size_t)row * N + col] = v;
    }
#if USE_ASYNC_LDS
    p ^= 1;
#endif
  }
}

// ---------------------------------------------------------------------------
// LayerNorm over 256 (wave per row, shuffle reductions)
// ---------------------------------------------------------------------------
template <bool OBF>
__global__ __launch_bounds__(256) void ln_kernel(
    const float* __restrict__ X, const float* __restrict__ g, const float* __restrict__ b,
    float* __restrict__ outF, bf16_t* __restrict__ outB) {
  const int w = threadIdx.x >> 5, lane = threadIdx.x & 31;
  const int row = blockIdx.x * 8 + w;
  const float* xr = X + (size_t)row * HH;
  float e[8]; float s = 0.f;
#pragma unroll
  for (int i = 0; i < 8; ++i) { e[i] = xr[lane + i * 32]; s += e[i]; }
#pragma unroll
  for (int off = 1; off < 32; off <<= 1) s += __shfl_xor(s, off, 32);
  const float m = s * (1.0f / 256.0f);
  float v = 0.f;
#pragma unroll
  for (int i = 0; i < 8; ++i) { float d = e[i] - m; v += d * d; }
#pragma unroll
  for (int off = 1; off < 32; off <<= 1) v += __shfl_xor(v, off, 32);
  const float rs = rsqrtf(v * (1.0f / 256.0f) + 1e-5f);
#pragma unroll
  for (int i = 0; i < 8; ++i) {
    const int c = lane + i * 32;
    float o = (e[i] - m) * rs * g[c] + b[c];
    if (OBF) outB[(size_t)row * HH + c] = f2bf(o);
    else     outF[(size_t)row * HH + c] = o;
  }
}

// ---------------------------------------------------------------------------
// Token MLP #1: x1[b,j,d] = sum_t quant[b,t,d] * w[j,t] + bias[j]  (bf16 out)
// ---------------------------------------------------------------------------
__global__ __launch_bounds__(256) void tokmlp_kernel(
    const float* __restrict__ Q, const float* __restrict__ W,
    const float* __restrict__ Bv, bf16_t* __restrict__ X1) {
  __shared__ float sw[JJ * TOKN];
  __shared__ float sbv[JJ];
  const int t = threadIdx.x;
  for (int i = t; i < JJ * TOKN; i += 256) sw[i] = W[i];
  if (t < JJ) sbv[t] = Bv[t];
  __syncthreads();

  const int bi = blockIdx.x >> 1;
  const int d = ((blockIdx.x & 1) << 8) + t;
  float acc[JJ];
#pragma unroll
  for (int j = 0; j < JJ; ++j) acc[j] = sbv[j];
  const float* qb = Q + (size_t)bi * TOKN * TD + d;
  for (int tt = 0; tt < TOKN; ++tt) {
    const float q = qb[(size_t)tt * TD];
#pragma unroll
    for (int j = 0; j < JJ; ++j) acc[j] += q * sw[j * TOKN + tt];
  }
  bf16_t* xb = X1 + (size_t)bi * JJ * TD + d;
#pragma unroll
  for (int j = 0; j < JJ; ++j) xb[(size_t)j * TD] = f2bf(acc[j]);
}

// ---------------------------------------------------------------------------
// Token mixing MLP (17 -> 64 -> 17 over j axis), writes r1 = x + y_mixed
// ---------------------------------------------------------------------------
__global__ __launch_bounds__(256) void tokmix_kernel(
    const float* __restrict__ Y, const float* __restrict__ X,
    const float* __restrict__ W1, const float* __restrict__ B1,
    const float* __restrict__ W2, const float* __restrict__ B2,
    float* __restrict__ R) {
  __shared__ float sw1[TI * JJ], sw2[JJ * TI], sb1[TI], sb2[JJ];
  const int t = threadIdx.x;
  for (int i = t; i < TI * JJ; i += 256) { sw1[i] = W1[i]; sw2[i] = W2[i]; }
  if (t < TI) sb1[t] = B1[t];
  if (t < JJ) sb2[t] = B2[t];
  __syncthreads();

  const int bi = blockIdx.x;
  const float* yb = Y + (size_t)bi * JJ * HH + t;
  float v[JJ];
#pragma unroll
  for (int j = 0; j < JJ; ++j) v[j] = yb[(size_t)j * HH];
  float y2[JJ];
#pragma unroll
  for (int j = 0; j < JJ; ++j) y2[j] = sb2[j];
  for (int ti = 0; ti < TI; ++ti) {
    float s = sb1[ti];
#pragma unroll
    for (int j = 0; j < JJ; ++j) s += v[j] * sw1[ti * JJ + j];
    const float gv = gelu_exact(s);
#pragma unroll
    for (int j = 0; j < JJ; ++j) y2[j] += gv * sw2[j * TI + ti];
  }
  const float* xb = X + (size_t)bi * JJ * HH + t;
  float* rb = R + (size_t)bi * JJ * HH + t;
#pragma unroll
  for (int j = 0; j < JJ; ++j) rb[(size_t)j * HH] = xb[(size_t)j * HH] + y2[j];
}

// ---------------------------------------------------------------------------
// Final LN fused with joints = xln @ rec_w^T + rec_b  (N=2)
// ---------------------------------------------------------------------------
__global__ __launch_bounds__(256) void final_kernel(
    const float* __restrict__ X, const float* __restrict__ g, const float* __restrict__ b,
    const float* __restrict__ rw, const float* __restrict__ rb,
    float* __restrict__ joints) {
  const int w = threadIdx.x >> 5, lane = threadIdx.x & 31;
  const int row = blockIdx.x * 8 + w;
  const float* xr = X + (size_t)row * HH;
  float e[8]; float s = 0.f;
#pragma unroll
  for (int i = 0; i < 8; ++i) { e[i] = xr[lane + i * 32]; s += e[i]; }
#pragma unroll
  for (int off = 1; off < 32; off <<= 1) s += __shfl_xor(s, off, 32);
  const float m = s * (1.0f / 256.0f);
  float v = 0.f;
#pragma unroll
  for (int i = 0; i < 8; ++i) { float d = e[i] - m; v += d * d; }
#pragma unroll
  for (int off = 1; off < 32; off <<= 1) v += __shfl_xor(v, off, 32);
  const float rs = rsqrtf(v * (1.0f / 256.0f) + 1e-5f);
  float d0 = 0.f, d1 = 0.f;
#pragma unroll
  for (int i = 0; i < 8; ++i) {
    const int c = lane + i * 32;
    const float o = (e[i] - m) * rs * g[c] + b[c];
    d0 += o * rw[c];
    d1 += o * rw[HH + c];
  }
#pragma unroll
  for (int off = 1; off < 32; off <<= 1) {
    d0 += __shfl_xor(d0, off, 32);
    d1 += __shfl_xor(d1, off, 32);
  }
  if (lane == 0) {
    joints[(size_t)row * 2 + 0] = d0 + rb[0];
    joints[(size_t)row * 2 + 1] = d1 + rb[1];
  }
}

// ---------------------------------------------------------------------------
// Host launcher
// ---------------------------------------------------------------------------
extern "C" void kernel_launch(void* const* d_in, const int* in_sizes, int n_in,
                              void* d_out, int out_size, void* d_ws, size_t ws_size,
                              hipStream_t stream) {
  (void)in_sizes; (void)n_in; (void)out_size; (void)ws_size;
  const float* E      = (const float*)d_in[0];
  const float* CB     = (const float*)d_in[1];
  const float* TMW    = (const float*)d_in[2];
  const float* TMB    = (const float*)d_in[3];
  const float* DSW    = (const float*)d_in[4];
  const float* DSB    = (const float*)d_in[5];
  const float* LN1G   = (const float*)d_in[6];
  const float* LN1B   = (const float*)d_in[7];
  const float* TW1    = (const float*)d_in[8];
  const float* TB1    = (const float*)d_in[9];
  const float* TW2    = (const float*)d_in[10];
  const float* TB2    = (const float*)d_in[11];
  const float* LN2G   = (const float*)d_in[12];
  const float* LN2B   = (const float*)d_in[13];
  const float* CW1    = (const float*)d_in[14];
  const float* CB1    = (const float*)d_in[15];
  const float* CW2    = (const float*)d_in[16];
  const float* CB2    = (const float*)d_in[17];
  const float* LNFG   = (const float*)d_in[18];
  const float* LNFB   = (const float*)d_in[19];
  const float* RECW   = (const float*)d_in[20];
  const float* RECB   = (const float*)d_in[21];

  // d_out sections (floats), concatenated in return order
  float* out    = (float*)d_out;
  float* orig   = out;                                   // 69632*512
  float* enc    = out + (size_t)NE * TD;                 // 69632*2048
  float* idxo   = enc + (size_t)NE * KCB;                // 69632
  float* joints = idxo + NE;                             // 2048*17*2

  // workspace layout (bytes)
  char* ws = (char*)d_ws;
  bf16_t* cb_bf = (bf16_t*)(ws + 0);             // 2,097,152
  float*  cnorm = (float*) (ws + 2097152);       // 8,192
  bf16_t* dsw   = (bf16_t*)(ws + 2105344);       // 262,144
  bf16_t* chw1  = (bf16_t*)(ws + 2367488);       // 1,048,576
  bf16_t* chw2  = (bf16_t*)(ws + 3416064);       // 1,048,576
  bf16_t* x1h1  = (bf16_t*)(ws + 4464640);       // 35,651,584  (x1 then reused as h1)
  float*  xf    = (float*) (ws + 40116224);      // 35,651,584
  float*  yln   = (float*) (ws + 75767808);      // 35,651,584
  float*  r1    = (float*) (ws + 111419392);     // 35,651,584
  bf16_t* ubf   = (bf16_t*)(ws + 147070976);     // 17,825,792

  // 1) zero the one-hot output (float4 stores)
  zero4_kernel<<<4096, 256, 0, stream>>>(enc, (long long)NE * KCB / 4);

  // 2) precompute bf16 copies + codebook norms
  cvt_bf16_kernel<<<512, 256, 0, stream>>>(CB, cb_bf, KCB * TD);
  cnorm_kernel<<<KCB / 8, 256, 0, stream>>>(CB, cnorm);
  cvt_bf16_kernel<<<128, 256, 0, stream>>>(DSW, dsw, HH * TD);
  cvt_bf16_kernel<<<256, 256, 0, stream>>>(CW1, chw1, LL * HID * HH);
  cvt_bf16_kernel<<<256, 256, 0, stream>>>(CW2, chw2, LL * HH * HID);

  // 3) quantization: WMMA distances + argmin + gather + one-hot scatter
  quant_kernel<<<NE / 128, 256, 0, stream>>>(E, cb_bf, cnorm, CB, orig, enc, idxo);

  // 4) token MLP over token dim (34 -> 17), bf16 output
  tokmlp_kernel<<<BS * 2, 256, 0, stream>>>(orig, TMW, TMB, x1h1);

  // 5) dec_start GEMM: [34816,512] x [256,512]^T -> xf
  gemm_wmma_kernel<512, false, false, false><<<MROWS / 128, 256, 0, stream>>>(
      x1h1, dsw, DSB, nullptr, xf, nullptr, HH);

  // 6) mixer layers
  for (int i = 0; i < LL; ++i) {
    ln_kernel<false><<<MROWS / 8, 256, 0, stream>>>(
        xf, LN1G + i * HH, LN1B + i * HH, yln, nullptr);
    tokmix_kernel<<<BS, 256, 0, stream>>>(
        yln, xf, TW1 + i * TI * JJ, TB1 + i * TI, TW2 + i * JJ * TI, TB2 + i * JJ, r1);
    ln_kernel<true><<<MROWS / 8, 256, 0, stream>>>(
        r1, LN2G + i * HH, LN2B + i * HH, nullptr, ubf);
    // channel MLP: h1 = gelu(u @ chw1^T + b1)  [34816,512] bf16
    gemm_wmma_kernel<256, true, false, true><<<MROWS / 128, 256, 0, stream>>>(
        ubf, chw1 + (size_t)i * HID * HH, CB1 + i * HID, nullptr, nullptr, x1h1, HID);
    // x = r1 + h1 @ chw2^T + b2   [34816,256] f32
    gemm_wmma_kernel<512, false, true, false><<<MROWS / 128, 256, 0, stream>>>(
        x1h1, chw2 + (size_t)i * HH * HID, CB2 + i * HH, r1, xf, nullptr, HH);
  }

  // 7) final LN fused with joints projection
  final_kernel<<<MROWS / 8, 256, 0, stream>>>(xf, LNFG, LNFB, RECW, RECB, joints);
}
